// FusedStaticNeRF_79190607004208
// MI455X (gfx1250) — matrix operand order repaired and checked by reading, not compiled, not run on hardware
//
#include <hip/hip_runtime.h>

typedef __attribute__((ext_vector_type(16))) _Float16 v16h;
typedef __attribute__((ext_vector_type(8)))  _Float16 v8h;
typedef __attribute__((ext_vector_type(8)))  float    v8f;

#define PI_F 3.14159265358979323846f

// ---------------- LDS geometry (units: halves unless noted) ----------------
// Activation tile per wave: 32 rows x SA halves (two 16-row M-tiles).
// SA padded 192->200 so 16 A-fragment lanes (stride 100 dwords) hit 16
// distinct 4-dword bank groups.
constexpr int SA     = 200;
constexpr int ACT_H  = 8 * 32 * SA;          // 51200 halves (8 waves x 32 rows)

// Weight arena (reused across two phases). K strides padded to odd*8 halves
// for bank-conflict-free B-fragment loads; 16B aligned offsets.
constexpr int WAR_H  = 103680;
// phase A (MLP1 weights)
constexpr int W10_OFF = 0;       constexpr int SK10 = 72;   // 128 x 72  (K=64 logical)
constexpr int W11_OFF = 9216;    constexpr int SK11 = 136;  // 128 x 136 (K=128)
constexpr int W12_OFF = 26624;                              // 128 x 136
// phase B (MLP2 + color weights)
constexpr int W20_OFF = 0;       constexpr int SK20 = 200;  // 128 x 200 (K=192 logical)
constexpr int W21_OFF = 25600;                              // 128 x 136
constexpr int W22_OFF = 43008;                              // 128 x 136
constexpr int W23_OFF = 60416;   constexpr int SK23 = 136;  // 144 x 136 (col-permuted 129-wide)
constexpr int WC0_OFF = 80000;   constexpr int SKC0 = 168;  // 128 x 168 (K=160 logical)
constexpr int WC1_OFF = 101504;  constexpr int SKC1 = 136;  // 16  x 136

constexpr size_t SH_BYTES = (size_t)(ACT_H + WAR_H) * 2 + (size_t)256 * 2 * 4; // 311,808 B

// ---------------------------------------------------------------------------
union V16U { v16h v; v8h p[2]; };

// A fragment (16x32 f16): lane r=lane&15 holds row r; lo lanes K=[kc..kc+7]
// then [kc+16..kc+23]; hi lanes +8. Two contiguous 16B LDS loads.
__device__ __forceinline__ v16h loadAfrag(const _Float16* actRow, int kc, int hh) {
  V16U u;
  u.p[0] = *(const v8h*)(actRow + kc + hh * 8);
  u.p[1] = *(const v8h*)(actRow + kc + 16 + hh * 8);
  return u.v;
}

// B fragment (32x16 f16) from transposed weights Wt[n][k]: lane holds column
// n=lane&15, lo lanes K=[kc..kc+7] then [kc+16..kc+23]; hi lanes +8.
__device__ __forceinline__ v16h loadBfrag(const _Float16* Wt, int SK, int n, int kc, int hh) {
  const _Float16* p = Wt + n * SK + kc + hh * 8;
  V16U u;
  u.p[0] = *(const v8h*)(p);
  u.p[1] = *(const v8h*)(p + 16);
  return u.v;
}

// One layer for TWO 16-row tiles sharing every B fragment (halves B traffic):
// D{0,1}(16 x NT*16) = A{0,1}(16 x KC*32) * Wt
template <int KC, int NT>
__device__ __forceinline__ void doLayer2(const _Float16* Arow0, const _Float16* Arow1,
                                         int kbase, const _Float16* Wt, int SK,
                                         int lane, v8f* acc0, v8f* acc1) {
  const int n  = lane & 15;
  const int hh = lane >> 4;
  v8f z = {};
#pragma unroll
  for (int t = 0; t < NT; ++t) { acc0[t] = z; acc1[t] = z; }
#pragma unroll
  for (int c = 0; c < KC; ++c) {
    v16h a0 = loadAfrag(Arow0, kbase + c * 32, hh);
    v16h a1 = loadAfrag(Arow1, kbase + c * 32, hh);
#pragma unroll
    for (int t = 0; t < NT; ++t) {
      v16h b = loadBfrag(Wt, SK, t * 16 + n, c * 32, hh);
      acc0[t] = __builtin_amdgcn_wmma_f32_16x16x32_f16(
          false, a0, false, b, (short)0, acc0[t], false, false);
      acc1[t] = __builtin_amdgcn_wmma_f32_16x16x32_f16(
          false, a1, false, b, (short)0, acc1[t], false, false);
    }
  }
}

// Store D tiles back to one 16-row activation tile (f16), optional ReLU.
// D layout: VGPR j -> row hh*8+j, column = t*16 + (lane&15).
template <int NT, bool RELU>
__device__ __forceinline__ void storeAct(_Float16* AwM, int colbase, int lane, const v8f* acc) {
  const int n  = lane & 15;
  const int hh = lane >> 4;
#pragma unroll
  for (int t = 0; t < NT; ++t) {
#pragma unroll
    for (int j = 0; j < 8; ++j) {
      float v = acc[t][j];
      if (RELU) v = v > 0.f ? v : 0.f;
      AwM[(hh * 8 + j) * SA + colbase + t * 16 + n] = (_Float16)v;
    }
  }
}

__global__ __launch_bounds__(256)
void fused_nerf_kernel(const float* __restrict__ x,
                       const float* __restrict__ W10, const float* __restrict__ W11,
                       const float* __restrict__ W12, const float* __restrict__ W20,
                       const float* __restrict__ W21, const float* __restrict__ W22,
                       const float* __restrict__ W23, const float* __restrict__ Wc0,
                       const float* __restrict__ Wc1,
                       float* __restrict__ out, int nrows) {
  extern __shared__ char smemRaw[];
  _Float16* act    = (_Float16*)smemRaw;
  _Float16* warena = act + ACT_H;
  float*    fscr   = (float*)(smemRaw + (size_t)(ACT_H + WAR_H) * 2); // [256][2] = {blend, density}

  const int tid  = threadIdx.x;
  const int wave = tid >> 5;
  const int lane = tid & 31;
  const int r    = lane & 15;
  const int hh   = lane >> 4;
  _Float16* Aw0 = act + wave * 32 * SA;            // M-tile 0 (rows 0..15)
  _Float16* Aw1 = Aw0 + 16 * SA;                   // M-tile 1 (rows 16..31)
  const _Float16* Arow0 = Aw0 + r * SA;
  const _Float16* Arow1 = Aw1 + r * SA;

  // ---------------- Phase A weight load (MLP1), f16 transposed -------------
  for (int i = tid; i < 128 * SK10; i += 256) {            // W10t: K=60 real, pad->72
    int n = i / SK10, k = i % SK10;
    warena[W10_OFF + i] = (k < 60) ? (_Float16)W10[k * 128 + n] : (_Float16)0.f;
  }
  for (int i = tid; i < 128 * SK11; i += 256) {            // W11t
    int n = i / SK11, k = i % SK11;
    warena[W11_OFF + i] = (k < 128) ? (_Float16)W11[k * 128 + n] : (_Float16)0.f;
  }
  for (int i = tid; i < 128 * SK11; i += 256) {            // W12t
    int n = i / SK11, k = i % SK11;
    warena[W12_OFF + i] = (k < 128) ? (_Float16)W12[k * 128 + n] : (_Float16)0.f;
  }

  // ---------------- Positional encoding (both M-tiles) ---------------------
  long long gbase = (long long)blockIdx.x * 256 + wave * 32 + r;
  float px[2][3], vx[2][3];
#pragma unroll
  for (int m = 0; m < 2; ++m) {
    long long g  = gbase + m * 16;
    long long gc = g < nrows ? g : (long long)nrows - 1;
#pragma unroll
    for (int d = 0; d < 3; ++d) { px[m][d] = x[gc * 6 + d]; vx[m][d] = x[gc * 6 + 3 + d]; }
  }
#pragma unroll
  for (int m = 0; m < 2; ++m) {
    _Float16* AwM = m ? Aw1 : Aw0;
#pragma unroll
    for (int i = 0; i < 32; ++i) {                          // f = dim*20 + freq*2 + {sin,cos}
      int f = hh * 32 + i;
      _Float16 v = (_Float16)0.f;
      if (f < 60) {
        int dim = f / 20, rem = f % 20, fr = rem >> 1;
        float ang = px[m][dim] * (PI_F * (float)(1 << fr));
        v = (_Float16)((rem & 1) ? __cosf(ang) : __sinf(ang));
      }
      AwM[r * SA + f] = v;
    }
  }
  __syncthreads();

  v8f acc0[9], acc1[9];
  // ---------------- MLP1: 64->128 relu, 128->128 relu, 128->128 -----------
  doLayer2<2, 8>(Arow0, Arow1, 0,  warena + W10_OFF, SK10, lane, acc0, acc1);
  storeAct<8, true >(Aw0, 64, lane, acc0);
  storeAct<8, true >(Aw1, 64, lane, acc1);
  doLayer2<4, 8>(Arow0, Arow1, 64, warena + W11_OFF, SK11, lane, acc0, acc1);
  storeAct<8, true >(Aw0, 64, lane, acc0);
  storeAct<8, true >(Aw1, 64, lane, acc1);
  doLayer2<4, 8>(Arow0, Arow1, 64, warena + W12_OFF, SK11, lane, acc0, acc1);
  storeAct<8, false>(Aw0, 64, lane, acc0);                  // part1 -> cols 64..191
  storeAct<8, false>(Aw1, 64, lane, acc1);
  __syncthreads();

  // ---------------- Phase B weight load (MLP2 + color) ---------------------
  for (int i = tid; i < 128 * SK20; i += 256) {            // W20t: [enc60|pad4|part1 128]
    int n = i / SK20, k = i % SK20;
    _Float16 v = (_Float16)0.f;
    if (k < 60)       v = (_Float16)W20[k * 128 + n];
    else if (k >= 64 && k < 192) v = (_Float16)W20[(k - 4) * 128 + n];
    warena[W20_OFF + i] = v;
  }
  for (int i = tid; i < 128 * SK11; i += 256) {            // W21t
    int n = i / SK11, k = i % SK11;
    warena[W21_OFF + i] = (k < 128) ? (_Float16)W21[k * 128 + n] : (_Float16)0.f;
  }
  for (int i = tid; i < 128 * SK11; i += 256) {            // W22t
    int n = i / SK11, k = i % SK11;
    warena[W22_OFF + i] = (k < 128) ? (_Float16)W22[k * 128 + n] : (_Float16)0.f;
  }
  for (int i = tid; i < 144 * SK23; i += 256) {            // W23t, col-permute: src col0 -> slot 128
    int n = i / SK23, k = i % SK23;
    _Float16 v = (_Float16)0.f;
    if (k < 128) {
      if (n < 128)       v = (_Float16)W23[k * 129 + n + 1];    // feature cols
      else if (n == 128) v = (_Float16)W23[k * 129];            // blending logit col
    }
    warena[W23_OFF + i] = v;
  }
  for (int i = tid; i < 128 * SKC0; i += 256) {            // Wc0t: [encview24|pad8|feature 128]
    int n = i / SKC0, k = i % SKC0;
    _Float16 v = (_Float16)0.f;
    if (k < 24)       v = (_Float16)Wc0[k * 128 + n];
    else if (k >= 32 && k < 160) v = (_Float16)Wc0[(k - 8) * 128 + n];
    warena[WC0_OFF + i] = v;
  }
  for (int i = tid; i < 16 * SKC1; i += 256) {             // Wc1t (3 real cols)
    int n = i / SKC1, k = i % SKC1;
    warena[WC1_OFF + i] = (n < 3 && k < 128) ? (_Float16)Wc1[k * 3 + n] : (_Float16)0.f;
  }
  __syncthreads();

  // ---------------- MLP2: 192->128 relu, 128 relu, 128 relu, 128->144 -----
  doLayer2<6, 8>(Arow0, Arow1, 0, warena + W20_OFF, SK20, lane, acc0, acc1);
  storeAct<8, true>(Aw0, 0, lane, acc0);
  storeAct<8, true>(Aw1, 0, lane, acc1);
  doLayer2<4, 8>(Arow0, Arow1, 0, warena + W21_OFF, SK11, lane, acc0, acc1);
  storeAct<8, true>(Aw0, 0, lane, acc0);
  storeAct<8, true>(Aw1, 0, lane, acc1);
  doLayer2<4, 8>(Arow0, Arow1, 0, warena + W22_OFF, SK11, lane, acc0, acc1);
  storeAct<8, true>(Aw0, 0, lane, acc0);
  storeAct<8, true>(Aw1, 0, lane, acc1);
  doLayer2<4, 9>(Arow0, Arow1, 0, warena + W23_OFF, SK23, lane, acc0, acc1);
  storeAct<8, false>(Aw0, 32, lane, acc0);                  // feature -> cols 32..159
  storeAct<8, false>(Aw1, 32, lane, acc1);
  if ((lane & 15) == 0) {                                    // tile8 col0 = blending, tile0 col0 = density
#pragma unroll
    for (int j = 0; j < 8; ++j) {
      int row0 = wave * 32 + hh * 8 + j;
      fscr[row0 * 2 + 0] = 1.f / (1.f + __expf(-acc0[8][j]));
      fscr[row0 * 2 + 1] = acc0[0][j];
      int row1 = row0 + 16;
      fscr[row1 * 2 + 0] = 1.f / (1.f + __expf(-acc1[8][j]));
      fscr[row1 * 2 + 1] = acc1[0][j];
    }
  }

  // ---------------- View encoding into cols 0..31 (both M-tiles) -----------
#pragma unroll
  for (int m = 0; m < 2; ++m) {
    _Float16* AwM = m ? Aw1 : Aw0;
#pragma unroll
    for (int i = 0; i < 16; ++i) {                           // f = dim*8 + freq*2 + {sin,cos}
      int f = hh * 16 + i;
      _Float16 v = (_Float16)0.f;
      if (f < 24) {
        int dim = f / 8, rem = f % 8, fr = rem >> 1;
        float ang = vx[m][dim] * (PI_F * (float)(1 << fr));
        v = (_Float16)((rem & 1) ? __cosf(ang) : __sinf(ang));
      }
      AwM[r * SA + f] = v;
    }
  }

  // ---------------- Color MLP: 160->128 relu, 128->3 ----------------------
  doLayer2<5, 8>(Arow0, Arow1, 0, warena + WC0_OFF, SKC0, lane, acc0, acc1);
  storeAct<8, true>(Aw0, 0, lane, acc0);
  storeAct<8, true>(Aw1, 0, lane, acc1);
  doLayer2<4, 1>(Arow0, Arow1, 0, warena + WC1_OFF, SKC1, lane, acc0, acc1);

  // ---------------- Output: [rgb0, rgb1, rgb2, density, blending] ----------
  const int n = lane & 15;
#pragma unroll
  for (int m = 0; m < 2; ++m) {
    const v8f& rgb = m ? acc1[0] : acc0[0];
    long long rowg0 = (long long)blockIdx.x * 256 + wave * 32 + m * 16 + hh * 8;
    int srow0 = wave * 32 + m * 16 + hh * 8;
    if (n < 3) {
#pragma unroll
      for (int j = 0; j < 8; ++j) {
        long long rr = rowg0 + j;
        if (rr < nrows) out[rr * 5 + n] = rgb[j];
      }
    } else if (n == 3) {
#pragma unroll
      for (int j = 0; j < 8; ++j) {
        long long rr = rowg0 + j;
        if (rr < nrows) out[rr * 5 + 3] = fscr[(srow0 + j) * 2 + 1];
      }
    } else if (n == 4) {
#pragma unroll
      for (int j = 0; j < 8; ++j) {
        long long rr = rowg0 + j;
        if (rr < nrows) out[rr * 5 + 4] = fscr[(srow0 + j) * 2 + 0];
      }
    }
  }
}

extern "C" void kernel_launch(void* const* d_in, const int* in_sizes, int n_in,
                              void* d_out, int out_size, void* d_ws, size_t ws_size,
                              hipStream_t stream) {
  const float* x   = (const float*)d_in[0];
  const float* W10 = (const float*)d_in[1];
  const float* W11 = (const float*)d_in[2];
  const float* W12 = (const float*)d_in[3];
  const float* W20 = (const float*)d_in[4];
  const float* W21 = (const float*)d_in[5];
  const float* W22 = (const float*)d_in[6];
  const float* W23 = (const float*)d_in[7];
  const float* Wc0 = (const float*)d_in[8];
  const float* Wc1 = (const float*)d_in[9];
  float* out = (float*)d_out;

  int nrows = in_sizes[0] / 6;
  int grid  = (nrows + 255) / 256;
  if (grid < 1) return;

  (void)hipFuncSetAttribute((const void*)fused_nerf_kernel,
                            hipFuncAttributeMaxDynamicSharedMemorySize,
                            (int)SH_BYTES);
  fused_nerf_kernel<<<grid, 256, SH_BYTES, stream>>>(
      x, W10, W11, W12, W20, W21, W22, W23, Wc0, Wc1, out, nrows);
}
